// GAT_62036507623870
// MI455X (gfx1250) — matrix operand order repaired and checked by reading, not compiled, not run on hardware
//
#include <hip/hip_runtime.h>
#include <hip/hip_bf16.h>

typedef __attribute__((ext_vector_type(16))) __bf16 v16bf;
typedef __attribute__((ext_vector_type(8)))  float  v8f;
typedef unsigned int uint32;

static constexpr int   NN    = 50000;
static constexpr int   EE    = 500000;
static constexpr int   NPAD  = 50016;          // 16-aligned row count for WMMA tiles
static constexpr float SLOPE = 0.2f;

#define DEV __device__ __forceinline__

DEV float lrelu(float v) { return v >= 0.f ? v : SLOPE * v; }
DEV float eluf (float v) { return v >  0.f ? v : __expf(v) - 1.f; }

// per-source attention logits, vectorized for H==4
template <int H>
DEV void load_el(const float* __restrict__ el, int s, float (&e)[H]) {
    if constexpr (H == 4) {
        float4 v = reinterpret_cast<const float4*>(el)[s];
        e[0] = v.x; e[1] = v.y; e[2] = v.z; e[3] = v.w;
    } else {
        e[0] = el[s];
    }
}

// ---------------------------------------------------------------- CSR build
__global__ void k_zero_i32(int* p, int n) {
    int i = blockIdx.x * blockDim.x + threadIdx.x;
    if (i < n) p[i] = 0;
}

__global__ void k_hist(const int* __restrict__ dst, int* __restrict__ cnt, int e) {
    int i = blockIdx.x * blockDim.x + threadIdx.x;
    if (i < e) atomicAdd(&cnt[dst[i]], 1);
}

// single-block chunked exclusive scan
__global__ void k_scan(const int* __restrict__ cnt, int* __restrict__ row_start, int n) {
    __shared__ int sh[1024];
    __shared__ int carry;
    if (threadIdx.x == 0) carry = 0;
    __syncthreads();
    for (int base = 0; base < n; base += 1024) {
        int i = base + (int)threadIdx.x;
        int v = (i < n) ? cnt[i] : 0;
        sh[threadIdx.x] = v;
        __syncthreads();
        for (int off = 1; off < 1024; off <<= 1) {
            int t = (threadIdx.x >= (unsigned)off) ? sh[threadIdx.x - off] : 0;
            __syncthreads();
            sh[threadIdx.x] += t;
            __syncthreads();
        }
        if (i < n) row_start[i] = carry + sh[threadIdx.x] - v;   // exclusive
        __syncthreads();
        if (threadIdx.x == 1023) carry += sh[1023];
        __syncthreads();
    }
    if (threadIdx.x == 0) row_start[n] = carry;
}

__global__ void k_scatter(const int* __restrict__ dst, const int* __restrict__ row_start,
                          int* __restrict__ cur, int* __restrict__ eidx, int e) {
    int i = blockIdx.x * blockDim.x + threadIdx.x;
    if (i < e) {
        int d = dst[i];
        int pos = atomicAdd(&cur[d], 1);
        eidx[row_start[d] + pos] = i;
    }
}

// ------------------------------------------------------- f32 -> bf16 converts
__global__ void k_cvt_x(const float* __restrict__ x, __bf16* __restrict__ actb) {
    int i = blockIdx.x * blockDim.x + threadIdx.x;
    if (i >= NPAD * 256) return;
    int row = i >> 8, col = i & 255;
    float v = (row < NN && col < 128) ? x[row * 128 + col] : 0.f;
    actb[i] = (__bf16)v;
}

// weight pack: K-pair (k, k+1) for one output column in adjacent bf16s:
// uint Wp[p*dout + col] = { W[2p,col], W[2p+1,col] }
__global__ void k_pack(const float* __restrict__ W, __bf16* __restrict__ Wp,
                       int din, int dout) {
    int idx = blockIdx.x * blockDim.x + threadIdx.x;
    int total = (din >> 1) * dout;
    if (idx >= total) return;
    int p = idx / dout, col = idx % dout;
    Wp[2 * idx]     = (__bf16)W[(2 * p)     * dout + col];
    Wp[2 * idx + 1] = (__bf16)W[(2 * p + 1) * dout + col];
}

// layer-3 combined [256 x 128] = [ W3 (256x64) | res_W3 (256x64) ]
__global__ void k_pack2(const float* __restrict__ W, const float* __restrict__ R,
                        __bf16* __restrict__ Wp) {
    int idx = blockIdx.x * blockDim.x + threadIdx.x;
    if (idx >= 128 * 128) return;
    int p = idx / 128, col = idx % 128;
    const float* S = (col < 64) ? W : R;
    int c = col & 63;
    Wp[2 * idx]     = (__bf16)S[(2 * p)     * 64 + c];
    Wp[2 * idx + 1] = (__bf16)S[(2 * p + 1) * 64 + c];
}

// --------------------------------------------------------------- WMMA GEMM
// One block = 16 rows x 128 cols = 8 waves, each owning one 16x16 tile.
// The packed B panel (DIN/2 x 128 uints, <=64KB) is staged into LDS once per
// block with GLOBAL_LOAD_ASYNC_TO_LDS_B128 (ASYNCcnt), then all waves build
// B fragments from LDS while streaming A from global (b128-merged).
template <int DIN, int DOUT>
__global__ __launch_bounds__(256) void k_gemm(const __bf16* __restrict__ A,
                                              const __bf16* __restrict__ Bp,
                                              float* __restrict__ C) {
    constexpr int PAIRS   = DIN / 2;            // K pairs
    constexpr int COLS    = 128;                // panel width
    constexpr int nPanels = DOUT / COLS;
    __shared__ __align__(16) uint32 ldsB[PAIRS * COLS];

    int bm      = blockIdx.x / nPanels;         // row tile
    int colBase = (blockIdx.x % nPanels) * COLS;

    // ---- async-stage the B panel into LDS
    {
        const uint32* Bg = (const uint32*)Bp;
        constexpr int CHUNKS = PAIRS * COLS / 4;          // 16B transfers
        for (int i = threadIdx.x; i < CHUNKS; i += 256) {
            int p = (i * 4) / COLS;                        // COLS % 4 == 0
            int c = (i * 4) % COLS;
            const uint32* g = Bg + (size_t)p * DOUT + colBase + c;
            uint32 lds_off = (uint32)(uintptr_t)(&ldsB[i * 4]);
            asm volatile("global_load_async_to_lds_b128 %0, %1, off"
                         :: "v"(lds_off), "v"((unsigned long long)(uintptr_t)g)
                         : "memory");
        }
        asm volatile("s_wait_asynccnt 0x0" ::: "memory");
    }
    __syncthreads();

    int wave = threadIdx.x >> 5;                // local 16-col tile
    int lane = threadIdx.x & 31;
    int half = lane >> 4, l15 = lane & 15;
    int row  = bm * 16 + l15;

    v8f c = {0.f, 0.f, 0.f, 0.f, 0.f, 0.f, 0.f, 0.f};
    const uint32* Arow = (const uint32*)(A + (size_t)row * DIN);   // bf16 pairs

    for (int k0 = 0; k0 < DIN; k0 += 32) {
        union { v16bf v; uint32 u[8]; } a, b;
#pragma unroll
        for (int j = 0; j < 8; ++j) {
            // A 16x32 bf16 layout (matches B): VGPR j -> K = half*16 + 2j
            int ka = k0 + (half << 4) + (j << 1);
            a.u[j] = Arow[ka >> 1];
        }
        int kb = k0 + (half << 4);  // B 32x16: lanes 0-15 K 0-15, lanes 16-31 K 16-31
#pragma unroll
        for (int j = 0; j < 8; ++j)
            b.u[j] = ldsB[((kb >> 1) + j) * COLS + wave * 16 + l15];

        c = __builtin_amdgcn_wmma_f32_16x16x32_bf16(
                false, a.v, false, b.v, (short)0, c, false, false);
    }
    // C/D layout: VGPR r, lanes 0-15 -> (M=r, N=lane); lanes 16-31 -> (M=8+r, N=lane-16)
    float* Cr = C + (size_t)(bm * 16 + half * 8) * DOUT + colBase + wave * 16 + l15;
#pragma unroll
    for (int r = 0; r < 8; ++r) Cr[(size_t)r * DOUT] = c[r];
}

// ------------------------------------------------- attention scores el / er
template <int H>
__global__ void k_scores(const float* __restrict__ feat, int fstride,
                         const float* __restrict__ al, const float* __restrict__ ar,
                         float* __restrict__ el, float* __restrict__ er) {
    constexpr int D = H * 64, CH = D / 32, LPH = 64 / CH;   // lanes per head
    int wid  = (blockIdx.x * blockDim.x + threadIdx.x) >> 5;
    int lane = threadIdx.x & 31;
    if (wid >= NN) return;
    const float* fr = feat + (size_t)wid * fstride + lane * CH;
    float pl = 0.f, pr = 0.f;
#pragma unroll
    for (int cc = 0; cc < CH; ++cc) {
        float f = fr[cc];
        int g = lane * CH + cc;
        pl += f * al[g];
        pr += f * ar[g];
    }
#pragma unroll
    for (int m = 1; m < LPH; m <<= 1) {
        pl += __shfl_xor(pl, m, 32);
        pr += __shfl_xor(pr, m, 32);
    }
    if ((lane & (LPH - 1)) == 0) {
        int h = (lane * CH) >> 6;
        el[(size_t)wid * H + h] = pl;
        er[(size_t)wid * H + h] = pr;
    }
}

// --------------------------------------- fused edge-softmax + aggregation
// One wave32 per destination node over its CSR in-edge list.
// MODE 0: h_out = elu(agg); also emit bf16 copy
// MODE 1: h_out = elu(agg + h_in) in place; also emit bf16 copy
// MODE 2: out   = agg + feat[:,64:128]  (layer-3 residual from fused GEMM)
template <int H, int MODE>
__global__ void k_agg(const int* __restrict__ row_start, const int* __restrict__ eidx,
                      const int* __restrict__ src,
                      const float* __restrict__ el, const float* __restrict__ er,
                      const float* __restrict__ feat, int fstride,
                      const float* h_in, float* h_out, __bf16* actb,
                      float* __restrict__ out) {
    constexpr int D = H * 64, CH = D / 32;
    int wid  = blockIdx.x * (blockDim.x >> 5) + (threadIdx.x >> 5);
    int lane = threadIdx.x & 31;
    if (wid >= NN) return;
    int beg = row_start[wid], end = row_start[wid + 1];

    float erh[H];
    load_el<H>(er, wid, erh);

    // pass 1: per-head max of leaky_relu(el[src] + er[dst])
    float mh[H];
#pragma unroll
    for (int h = 0; h < H; ++h) mh[h] = -1e30f;
    for (int e = beg + lane; e < end; e += 32) {
        int s = src[eidx[e]];
        float es[H]; load_el<H>(el, s, es);
#pragma unroll
        for (int h = 0; h < H; ++h) mh[h] = fmaxf(mh[h], lrelu(es[h] + erh[h]));
    }
#pragma unroll
    for (int m = 16; m >= 1; m >>= 1)
#pragma unroll
        for (int h = 0; h < H; ++h) mh[h] = fmaxf(mh[h], __shfl_xor(mh[h], m, 32));

    // pass 2: denominator
    float sh[H];
#pragma unroll
    for (int h = 0; h < H; ++h) sh[h] = 0.f;
    for (int e = beg + lane; e < end; e += 32) {
        int s = src[eidx[e]];
        float es[H]; load_el<H>(el, s, es);
#pragma unroll
        for (int h = 0; h < H; ++h) sh[h] += __expf(lrelu(es[h] + erh[h]) - mh[h]);
    }
#pragma unroll
    for (int m = 16; m >= 1; m >>= 1)
#pragma unroll
        for (int h = 0; h < H; ++h) sh[h] += __shfl_xor(sh[h], m, 32);
    float inv[H];
#pragma unroll
    for (int h = 0; h < H; ++h) inv[h] = (sh[h] > 0.f) ? 1.f / sh[h] : 0.f;

    // pass 3: weighted gather-accumulate; whole wave walks each edge together,
    // lane reads its CH-float slice as float4s -> coalesced 1KB b128 bursts
    float acc[CH];
#pragma unroll
    for (int cc = 0; cc < CH; ++cc) acc[cc] = 0.f;
    const int h0 = (lane * CH) >> 6;          // CH | 64 -> single head per lane
    for (int e = beg; e < end; ++e) {
        int s = src[eidx[e]];
        if (e + 1 < end) {
            int sn = src[eidx[e + 1]];
            __builtin_prefetch(&feat[(size_t)sn * fstride + lane * CH], 0, 0);
        }
        float es[H]; load_el<H>(el, s, es);
        float wv = __expf(lrelu(es[h0] + erh[h0]) - mh[h0]) * inv[h0];
        const float* fr = feat + (size_t)s * fstride + lane * CH;
        if constexpr (CH == 8) {
            const float4* fr4 = reinterpret_cast<const float4*>(fr);
            float4 f0 = fr4[0], f1 = fr4[1];
            acc[0] += wv * f0.x; acc[1] += wv * f0.y;
            acc[2] += wv * f0.z; acc[3] += wv * f0.w;
            acc[4] += wv * f1.x; acc[5] += wv * f1.y;
            acc[6] += wv * f1.z; acc[7] += wv * f1.w;
        } else {
            const float2* fr2 = reinterpret_cast<const float2*>(fr);
            float2 f0 = fr2[0];
            acc[0] += wv * f0.x; acc[1] += wv * f0.y;
        }
    }

    size_t base = (size_t)wid * D + lane * CH;
#pragma unroll
    for (int cc = 0; cc < CH; ++cc) {
        if (MODE == 0) {
            float v = eluf(acc[cc]);
            h_out[base + cc] = v;
            actb[base + cc]  = (__bf16)v;
        } else if (MODE == 1) {
            float v = eluf(acc[cc] + h_in[base + cc]);   // in-place safe: same-thread RMW
            h_out[base + cc] = v;
            actb[base + cc]  = (__bf16)v;
        } else { // MODE 2: H==1, residual lives in cols 64..127 of fused GEMM output
            out[(size_t)wid * 64 + lane * CH + cc] =
                acc[cc] + feat[(size_t)wid * fstride + 64 + lane * CH + cc];
        }
    }
}

// ------------------------------------------------------------------ driver
extern "C" void kernel_launch(void* const* d_in, const int* in_sizes, int n_in,
                              void* d_out, int out_size, void* d_ws, size_t ws_size,
                              hipStream_t stream) {
    const float* x     = (const float*)d_in[0];
    const float* W1    = (const float*)d_in[1];
    const float* al1   = (const float*)d_in[2];
    const float* ar1   = (const float*)d_in[3];
    const float* W2    = (const float*)d_in[4];
    const float* al2   = (const float*)d_in[5];
    const float* ar2   = (const float*)d_in[6];
    const float* W3    = (const float*)d_in[7];
    const float* al3   = (const float*)d_in[8];
    const float* ar3   = (const float*)d_in[9];
    const float* rW3   = (const float*)d_in[10];
    const int*   src   = (const int*)d_in[11];
    const int*   dst   = (const int*)d_in[12];
    float*       outp  = (float*)d_out;

    char* ws = (char*)d_ws;
    size_t off = 0;
    auto alloc = [&](size_t bytes) -> void* {
        void* p = ws + off;
        off = (off + bytes + 255) & ~(size_t)255;
        return p;
    };
    __bf16* actb = (__bf16*)alloc((size_t)NPAD * 256 * 2);
    __bf16* w1p  = (__bf16*)alloc((size_t)128 * 256 * 2);
    __bf16* w2p  = (__bf16*)alloc((size_t)256 * 256 * 2);
    __bf16* w3p  = (__bf16*)alloc((size_t)256 * 128 * 2);
    float*  feat = (float*)alloc((size_t)NPAD * 256 * 4);
    float*  hbuf = (float*)alloc((size_t)NPAD * 256 * 4);
    float*  el   = (float*)alloc((size_t)NPAD * 4 * 4);
    float*  er   = (float*)alloc((size_t)NPAD * 4 * 4);
    int* row_start = (int*)alloc((size_t)(NN + 1) * 4);
    int* cnt       = (int*)alloc((size_t)NN * 4);
    int* eidx      = (int*)alloc((size_t)EE * 4);

    const int T = 256;
    const int MT = NPAD / 16;                       // 3126 row tiles
    auto cdiv = [](long long a, long long b) { return (int)((a + b - 1) / b); };

    // ---- CSR by destination (shared by all three layers)
    k_zero_i32<<<cdiv(NN, T), T, 0, stream>>>(cnt, NN);
    k_hist<<<cdiv(EE, T), T, 0, stream>>>(dst, cnt, EE);
    k_scan<<<1, 1024, 0, stream>>>(cnt, row_start, NN);
    k_zero_i32<<<cdiv(NN, T), T, 0, stream>>>(cnt, NN);
    k_scatter<<<cdiv(EE, T), T, 0, stream>>>(dst, row_start, cnt, eidx, EE);

    // ---- weight packing (f32 -> packed bf16)
    k_pack<<<cdiv(64 * 256, T), T, 0, stream>>>(W1, w1p, 128, 256);
    k_pack<<<cdiv(128 * 256, T), T, 0, stream>>>(W2, w2p, 256, 256);
    k_pack2<<<cdiv(128 * 128, T), T, 0, stream>>>(W3, rW3, w3p);

    // ---- layer 1
    k_cvt_x<<<cdiv((long long)NPAD * 256, T), T, 0, stream>>>(x, actb);
    k_gemm<128, 256><<<MT * 2, T, 0, stream>>>(actb, w1p, feat);
    k_scores<4><<<cdiv((long long)NN * 32, T), T, 0, stream>>>(feat, 256, al1, ar1, el, er);
    k_agg<4, 0><<<cdiv((long long)NN * 32, T), T, 0, stream>>>(
        row_start, eidx, src, el, er, feat, 256, nullptr, hbuf, actb, nullptr);

    // ---- layer 2 (identity residual)
    k_gemm<256, 256><<<MT * 2, T, 0, stream>>>(actb, w2p, feat);
    k_scores<4><<<cdiv((long long)NN * 32, T), T, 0, stream>>>(feat, 256, al2, ar2, el, er);
    k_agg<4, 1><<<cdiv((long long)NN * 32, T), T, 0, stream>>>(
        row_start, eidx, src, el, er, feat, 256, hbuf, hbuf, actb, nullptr);

    // ---- layer 3 (fused feat | residual GEMM, H = 1)
    k_gemm<256, 128><<<MT, T, 0, stream>>>(actb, w3p, feat);
    k_scores<1><<<cdiv((long long)NN * 32, T), T, 0, stream>>>(feat, 128, al3, ar3, el, er);
    k_agg<1, 2><<<cdiv((long long)NN * 32, T), T, 0, stream>>>(
        row_start, eidx, src, el, er, feat, 128, nullptr, nullptr, nullptr, outp);
}
